// MambaQNetwork_52604759441447
// MI455X (gfx1250) — matrix operand order repaired and checked by reading, not compiled, not run on hardware
//
#include <hip/hip_runtime.h>
#include <hip/hip_bf16.h>
#include <math.h>

// ---------------------------------------------------------------------------
// Mamba Q-network on gfx1250 (CDNA5). Dense GEMMs: v_wmma_f32_16x16x32_f16
// fed by global_load_b128 (all K dims pre-padded to multiples of 32, so the
// inner loop is branch-free). Scan/conv/LN stay in f32 VALU for precision.
// ---------------------------------------------------------------------------

typedef __attribute__((ext_vector_type(16))) _Float16 v16h;
typedef __attribute__((ext_vector_type(8)))  float    v8f;

#define M_TOT   16384      // B * L
#define B_      16
#define L_      1024
#define H_      256
#define DI_     512
#define N_      64
#define DC_     4
#define DR_     16
#define DBLW_   144        // DR + 2N
#define KCAT_   96         // 80 padded to 96
#define KDT_    32         // 16 padded to 32

enum { ACT_NONE = 0, ACT_RELU = 1, ACT_SILU = 2, ACT_SOFTPLUS = 3 };

__device__ __forceinline__ float act_apply(float x, int act) {
    switch (act) {
    case ACT_RELU:     return x > 0.f ? x : 0.f;
    case ACT_SILU:     return x / (1.f + expf(-x));
    case ACT_SOFTPLUS: return (x > 20.f) ? x : log1pf(expf(x));
    default:           return x;
    }
}

__device__ __forceinline__ void pack8(v16h& d, int o, float4 a, float4 b) {
    d[o + 0] = (_Float16)a.x; d[o + 1] = (_Float16)a.y;
    d[o + 2] = (_Float16)a.z; d[o + 3] = (_Float16)a.w;
    d[o + 4] = (_Float16)b.x; d[o + 5] = (_Float16)b.y;
    d[o + 6] = (_Float16)b.z; d[o + 7] = (_Float16)b.w;
}

// ---------------------------------------------------------------------------
// WMMA GEMM: C[M,Nc] = act( A[M,K] @ W[Nc,K]^T + bias ),  K % 32 == 0.
// One wave computes a 16 x (16*NT) tile; A fragment reused across NT WMMAs.
// grid = (M/16, Nc/(16*NT)), block = 32.
// ---------------------------------------------------------------------------
template <int NT>
__global__ void gemm_wmma_kernel(const float* __restrict__ A, int lda,
                                 const float* __restrict__ W, int ldw,
                                 const float* __restrict__ bias,
                                 float* __restrict__ C, int ldc,
                                 int K, int act) {
    const int row0 = blockIdx.x * 16;
    const int col0 = blockIdx.y * (16 * NT);
    const int lane = threadIdx.x & 31;
    const int lh   = lane & 15;
    const bool hi  = lane >= 16;
    const int koff  = hi ? 16 : 0;   // A-fragment K sub-offset per half-wave
    const int kboff = hi ? 16 : 0;   // B-fragment K sub-offset per half-wave

    v8f acc[NT];
    #pragma unroll
    for (int t = 0; t < NT; ++t) acc[t] = (v8f){};

    const float* __restrict__ arow = A + (size_t)(row0 + lh) * (size_t)lda;

    for (int k0 = 0; k0 < K; k0 += 32) {
        __builtin_prefetch(arow + k0 + 32, 0, 0);
        // A fragment: lanes 0-15 hold K k0..k0+15 contiguous,
        // lanes 16-31 hold K k0+16..k0+31. Four b128 loads.
        v16h a;
        {
            const float4* p0 = (const float4*)(arow + k0 + koff);
            pack8(a, 0, p0[0], p0[1]);
            pack8(a, 8, p0[2], p0[3]);
        }
        #pragma unroll
        for (int t = 0; t < NT; ++t) {
            // B fragment: B[k][n] = W[col+n][k0+k]; lanes 0-15 hold K k0..k0+15
            // contiguous, lanes 16-31 hold K k0+16..k0+31. Four b128 loads.
            const float4* q = (const float4*)(W +
                (size_t)(col0 + t * 16 + lh) * (size_t)ldw + k0 + kboff);
            v16h b;
            pack8(b, 0, q[0], q[1]);
            pack8(b, 8, q[2], q[3]);
            acc[t] = __builtin_amdgcn_wmma_f32_16x16x32_f16(
                false, a, false, b, (short)0, acc[t], false, false);
        }
    }

    #pragma unroll
    for (int t = 0; t < NT; ++t) {
        const int col = col0 + t * 16 + lh;
        const float bv = bias ? bias[col] : 0.f;
        #pragma unroll
        for (int v = 0; v < 8; ++v) {
            const int m = hi ? v + 8 : v;   // C/D layout: VGPR v -> row v / v+8
            C[(size_t)(row0 + m) * (size_t)ldc + col] =
                act_apply(acc[t][v] + bv, act);
        }
    }
}

// Zero-pad weight rows: dst[r,0:lds) = src[r,:], dst[r,lds:ldd) = 0.
__global__ void pad_rows_kernel(const float* __restrict__ src, int lds,
                                float* __restrict__ dst, int ldd, int rows) {
    int idx = blockIdx.x * blockDim.x + threadIdx.x;
    if (idx >= rows * ldd) return;
    int r = idx / ldd, k = idx % ldd;
    dst[idx] = (k < lds) ? src[r * lds + k] : 0.f;
}

// cat1 = [state|action|0pad], cat2 = [state|last_action|0pad]  (width 96)
__global__ void pack_cat_kernel(const float* __restrict__ state,
                                const float* __restrict__ action,
                                const float* __restrict__ last_action,
                                float* __restrict__ cat1,
                                float* __restrict__ cat2) {
    size_t idx = (size_t)blockIdx.x * blockDim.x + threadIdx.x;
    if (idx >= (size_t)M_TOT * KCAT_) return;
    size_t m = idx / KCAT_;
    int c = (int)(idx % KCAT_);
    float v1, v2;
    if (c < 64)       { float s = state[m * 64 + c]; v1 = s; v2 = s; }
    else if (c < 80)  { v1 = action[m * 16 + (c - 64)];
                        v2 = last_action[m * 16 + (c - 64)]; }
    else              { v1 = 0.f; v2 = 0.f; }
    cat1[idx] = v1;
    cat2[idx] = v2;
}

// dt GEMM input: dtin[m, 0:16) = dbl[m, 0:16), dtin[m, 16:32) = 0.
__global__ void pack_dtin_kernel(const float* __restrict__ dbl,
                                 float* __restrict__ dtin) {
    size_t idx = (size_t)blockIdx.x * blockDim.x + threadIdx.x;
    if (idx >= (size_t)M_TOT * KDT_) return;
    size_t m = idx / KDT_;
    int k = (int)(idx % KDT_);
    dtin[idx] = (k < DR_) ? dbl[m * DBLW_ + k] : 0.f;
}

// Depthwise causal conv1d (DC=4) over L, then SiLU. u = xz cols [0,512).
__global__ void conv_silu_kernel(const float* __restrict__ xz,
                                 const float* __restrict__ conv_w,
                                 const float* __restrict__ conv_b,
                                 float* __restrict__ uconv) {
    size_t idx = (size_t)blockIdx.x * blockDim.x + threadIdx.x;
    if (idx >= (size_t)M_TOT * DI_) return;
    size_t m = idx / DI_;
    int d = (int)(idx % DI_);
    int b = (int)(m / L_);
    int l = (int)(m % L_);
    float s = conv_b[d];
    #pragma unroll
    for (int k = 0; k < DC_; ++k) {
        int ls = l - (DC_ - 1) + k;
        if (ls >= 0)
            s += xz[((size_t)b * L_ + ls) * 1024 + d] * conv_w[d * DC_ + k];
    }
    uconv[idx] = s / (1.f + expf(-s));
}

// Selective scan: one wave per (b, d); lane n holds h[n], h[n+32].
__global__ void scan_kernel(const float* __restrict__ uconv,
                            const float* __restrict__ dtb,   // [M,1024] stride
                            const float* __restrict__ dbl,   // [M,144]
                            const float* __restrict__ A_log, // [512,64]
                            float* __restrict__ ys) {        // [M,512]
    const int wave = blockIdx.x * (blockDim.x >> 5) + (threadIdx.x >> 5);
    const int lane = threadIdx.x & 31;
    if (wave >= B_ * DI_) return;
    const int b = wave >> 9;
    const int d = wave & (DI_ - 1);

    const float A0 = -expf(A_log[d * N_ + lane]);
    const float A1 = -expf(A_log[d * N_ + lane + 32]);
    float h0 = 0.f, h1 = 0.f;

    for (int l = 0; l < L_; ++l) {
        const size_t row = (size_t)b * L_ + l;
        const float uv  = uconv[row * DI_ + d];
        const float dtv = dtb[row * 1024 + d];
        const float* __restrict__ dr = dbl + row * DBLW_;
        const float Bv0 = dr[DR_ + lane];
        const float Bv1 = dr[DR_ + 32 + lane];
        const float Cv0 = dr[DR_ + N_ + lane];
        const float Cv1 = dr[DR_ + N_ + 32 + lane];
        const float du = dtv * uv;
        h0 = expf(dtv * A0) * h0 + du * Bv0;
        h1 = expf(dtv * A1) * h1 + du * Bv1;
        float y = h0 * Cv0 + h1 * Cv1;
        #pragma unroll
        for (int off = 16; off > 0; off >>= 1)
            y += __shfl_xor(y, off, 32);
        if (lane == 0) ys[row * DI_ + d] = y;
    }
}

// y = (ys + u*D) * silu(z);  z = xz cols [512,1024). In-place into ys.
__global__ void y_combine_kernel(float* __restrict__ ys,
                                 const float* __restrict__ uconv,
                                 const float* __restrict__ xz,
                                 const float* __restrict__ Dp) {
    size_t idx = (size_t)blockIdx.x * blockDim.x + threadIdx.x;
    if (idx >= (size_t)M_TOT * DI_) return;
    size_t m = idx / DI_;
    int d = (int)(idx % DI_);
    float y = ys[idx] + uconv[idx] * Dp[d];
    float z = xz[m * 1024 + DI_ + d];
    ys[idx] = y * (z / (1.f + expf(-z)));
}

// LayerNorm over 256 cols; one wave per row; writes to dst at column offset.
__global__ void ln_kernel(const float* __restrict__ src,
                          const float* __restrict__ g,
                          const float* __restrict__ bta,
                          float* __restrict__ dst, int dst_ld, int dst_off) {
    const int row = blockIdx.x;
    const int lane = threadIdx.x & 31;
    float x[8];
    float s = 0.f;
    #pragma unroll
    for (int i = 0; i < 8; ++i) {
        x[i] = src[(size_t)row * H_ + i * 32 + lane];
        s += x[i];
    }
    #pragma unroll
    for (int off = 16; off > 0; off >>= 1) s += __shfl_xor(s, off, 32);
    const float mean = s * (1.f / H_);
    float v = 0.f;
    #pragma unroll
    for (int i = 0; i < 8; ++i) { float dd = x[i] - mean; v += dd * dd; }
    #pragma unroll
    for (int off = 16; off > 0; off >>= 1) v += __shfl_xor(v, off, 32);
    const float inv = rsqrtf(v * (1.f / H_) + 1e-5f);
    #pragma unroll
    for (int i = 0; i < 8; ++i) {
        int c = i * 32 + lane;
        dst[(size_t)row * dst_ld + dst_off + c] = (x[i] - mean) * inv * g[c] + bta[c];
    }
}

// out[m] = dot(x3[m,:], W4) + b4; one wave per row.
__global__ void final_dot_kernel(const float* __restrict__ x3,
                                 const float* __restrict__ W4,
                                 const float* __restrict__ b4,
                                 float* __restrict__ out) {
    const int row = blockIdx.x;
    const int lane = threadIdx.x & 31;
    float s = 0.f;
    #pragma unroll
    for (int i = 0; i < 8; ++i) {
        int c = i * 32 + lane;
        s += x3[(size_t)row * H_ + c] * W4[c];
    }
    #pragma unroll
    for (int off = 16; off > 0; off >>= 1) s += __shfl_xor(s, off, 32);
    if (lane == 0) out[row] = s + b4[0];
}

__global__ void copy_kernel(const float* __restrict__ src, float* __restrict__ dst, int n) {
    int i = blockIdx.x * blockDim.x + threadIdx.x;
    if (i < n) dst[i] = src[i];
}

// ---------------------------------------------------------------------------
extern "C" void kernel_launch(void* const* d_in, const int* in_sizes, int n_in,
                              void* d_out, int out_size, void* d_ws, size_t ws_size,
                              hipStream_t stream) {
    const float* state       = (const float*)d_in[0];
    const float* action      = (const float*)d_in[1];
    const float* last_action = (const float*)d_in[2];
    const float* hidden_in   = (const float*)d_in[3];
    const float* W1 = (const float*)d_in[4];   const float* b1 = (const float*)d_in[5];
    const float* W2 = (const float*)d_in[6];   const float* b2 = (const float*)d_in[7];
    const float* W3 = (const float*)d_in[8];   const float* b3 = (const float*)d_in[9];
    const float* W4 = (const float*)d_in[10];  const float* b4 = (const float*)d_in[11];
    const float* ln1_g = (const float*)d_in[12]; const float* ln1_b = (const float*)d_in[13];
    const float* ln2_g = (const float*)d_in[14]; const float* ln2_b = (const float*)d_in[15];
    const float* in_proj_w = (const float*)d_in[16];
    const float* conv_w    = (const float*)d_in[17];
    const float* conv_b    = (const float*)d_in[18];
    const float* x_proj_w  = (const float*)d_in[19];
    const float* dt_proj_w = (const float*)d_in[20];
    const float* dt_proj_b = (const float*)d_in[21];
    const float* A_log     = (const float*)d_in[22];
    const float* Dp        = (const float*)d_in[23];
    const float* out_proj_w = (const float*)d_in[24];
    float* out = (float*)d_out;
    float* ws  = (float*)d_ws;

    // workspace layout (floats)
    size_t off = 0;
    const size_t o_cat1  = off; off += (size_t)M_TOT * KCAT_;   // [M,96]
    const size_t o_cat2  = off; off += (size_t)M_TOT * KCAT_;   // [M,96]
    const size_t o_fc    = off; off += (size_t)M_TOT * H_;      // [M,256]
    const size_t o_mh    = off; off += (size_t)M_TOT * H_;      // [M,256]; reused as dbl [M,144]
    const size_t o_xz    = off; off += (size_t)M_TOT * 1024;    // [M,1024]; cols 0-511 reused for dt
    const size_t o_uconv = off; off += (size_t)M_TOT * DI_;     // [M,512]
    const size_t o_ys    = off; off += (size_t)M_TOT * DI_;     // [M,512]; becomes y
    const size_t o_mamba = off; off += (size_t)M_TOT * H_;      // [M,256]
    const size_t o_cat3  = off; off += (size_t)M_TOT * DI_;     // [M,512]
    const size_t o_x3    = off; off += (size_t)M_TOT * H_;      // [M,256]
    const size_t o_dtin  = off; off += (size_t)M_TOT * KDT_;    // [M,32]
    const size_t o_w1p   = off; off += (size_t)H_ * KCAT_;      // [256,96]
    const size_t o_w2p   = off; off += (size_t)H_ * KCAT_;      // [256,96]
    const size_t o_dtwp  = off; off += (size_t)DI_ * KDT_;      // [512,32]
    if (ws_size < off * sizeof(float)) return;

    float* cat1  = ws + o_cat1;
    float* cat2  = ws + o_cat2;
    float* fc    = ws + o_fc;
    float* mh    = ws + o_mh;
    float* dbl   = ws + o_mh;      // alias (mh dead after in_proj)
    float* xz    = ws + o_xz;
    float* dtb   = ws + o_xz;      // dt written into xz cols [0,512), ldc=1024
    float* uconv = ws + o_uconv;
    float* ys    = ws + o_ys;
    float* mamba = ws + o_mamba;
    float* cat3  = ws + o_cat3;
    float* x3    = ws + o_x3;
    float* dtin  = ws + o_dtin;
    float* W1p   = ws + o_w1p;
    float* W2p   = ws + o_w2p;
    float* dtwp  = ws + o_dtwp;

    // 0. pad weights to K%32==0
    pad_rows_kernel<<<(H_ * KCAT_ + 255) / 256, 256, 0, stream>>>(W1, 80, W1p, KCAT_, H_);
    pad_rows_kernel<<<(H_ * KCAT_ + 255) / 256, 256, 0, stream>>>(W2, 80, W2p, KCAT_, H_);
    pad_rows_kernel<<<(DI_ * KDT_ + 255) / 256, 256, 0, stream>>>(dt_proj_w, DR_, dtwp, KDT_, DI_);

    // 1. pack concat inputs (width 96, zero padded)
    {
        size_t n = (size_t)M_TOT * KCAT_;
        pack_cat_kernel<<<(unsigned)((n + 255) / 256), 256, 0, stream>>>(
            state, action, last_action, cat1, cat2);
    }
    // 2. fc = relu(cat1 @ W1^T + b1); mh = silu(cat2 @ W2^T + b2)
    gemm_wmma_kernel<4><<<dim3(M_TOT / 16, H_ / 64), 32, 0, stream>>>(
        cat1, KCAT_, W1p, KCAT_, b1, fc, H_, KCAT_, ACT_RELU);
    gemm_wmma_kernel<4><<<dim3(M_TOT / 16, H_ / 64), 32, 0, stream>>>(
        cat2, KCAT_, W2p, KCAT_, b2, mh, H_, KCAT_, ACT_SILU);
    // 3. xz = mh @ in_proj_w^T  [M,1024]
    gemm_wmma_kernel<4><<<dim3(M_TOT / 16, 1024 / 64), 32, 0, stream>>>(
        mh, H_, in_proj_w, H_, nullptr, xz, 1024, H_, ACT_NONE);
    // 4. depthwise causal conv + silu -> uconv
    {
        size_t n = (size_t)M_TOT * DI_;
        conv_silu_kernel<<<(unsigned)((n + 255) / 256), 256, 0, stream>>>(
            xz, conv_w, conv_b, uconv);
    }
    // 5. dbl = uconv @ x_proj_w^T  [M,144]  (overwrites mh buffer)
    gemm_wmma_kernel<1><<<dim3(M_TOT / 16, DBLW_ / 16), 32, 0, stream>>>(
        uconv, DI_, x_proj_w, DI_, nullptr, dbl, DBLW_, DI_, ACT_NONE);
    // 6. dt input pack (zero-padded K=32), then
    //    dt = softplus(dtin @ dtwp^T + dt_proj_b) -> xz cols [0,512)
    {
        size_t n = (size_t)M_TOT * KDT_;
        pack_dtin_kernel<<<(unsigned)((n + 255) / 256), 256, 0, stream>>>(dbl, dtin);
    }
    gemm_wmma_kernel<4><<<dim3(M_TOT / 16, DI_ / 64), 32, 0, stream>>>(
        dtin, KDT_, dtwp, KDT_, dt_proj_b, dtb, 1024, KDT_, ACT_SOFTPLUS);
    // 7. selective scan -> ys
    scan_kernel<<<(B_ * DI_) / 8, 256, 0, stream>>>(uconv, dtb, dbl, A_log, ys);
    // 8. y = (ys + u*D) * silu(z)  (in-place in ys)
    {
        size_t n = (size_t)M_TOT * DI_;
        y_combine_kernel<<<(unsigned)((n + 255) / 256), 256, 0, stream>>>(
            ys, uconv, xz, Dp);
    }
    // 9. mamba_out = y @ out_proj_w^T  [M,256]
    gemm_wmma_kernel<4><<<dim3(M_TOT / 16, H_ / 64), 32, 0, stream>>>(
        ys, DI_, out_proj_w, DI_, nullptr, mamba, H_, DI_, ACT_NONE);
    // 10. layernorms into cat3 = [ln2(fc) | ln1(mamba_out)]
    ln_kernel<<<M_TOT, 32, 0, stream>>>(mamba, ln1_g, ln1_b, cat3, DI_, H_);
    ln_kernel<<<M_TOT, 32, 0, stream>>>(fc,    ln2_g, ln2_b, cat3, DI_, 0);
    // 11. x3 = relu(cat3 @ W3^T + b3)  [M,256]
    gemm_wmma_kernel<4><<<dim3(M_TOT / 16, H_ / 64), 32, 0, stream>>>(
        cat3, DI_, W3, DI_, b3, x3, H_, DI_, ACT_RELU);
    // 12. Q = x3 @ W4^T + b4  -> out[0:16384]
    final_dot_kernel<<<M_TOT, 32, 0, stream>>>(x3, W4, b4, out);
    // 13. hidden passthrough -> out[16384:20480]
    copy_kernel<<<(4096 + 255) / 256, 256, 0, stream>>>(hidden_in, out + M_TOT, 4096);
}
